// StandardLayer_70566312673290
// MI455X (gfx1250) — matrix operand order repaired and checked by reading, not compiled, not run on hardware
//
#include <hip/hip_runtime.h>
#include <hip/hip_bf16.h>

// ---------------------------------------------------------------------------
// GRU layer for MI455X (gfx1250): bf16 WMMA, fragment-swizzled operands,
// persistent recurrent kernel, Wh cached in LDS, async global->LDS staging
// of h_{t-1}, grid-wide atomic barrier.
// ---------------------------------------------------------------------------

#define T_STEPS 512
#define BATCH   64
#define D_IN    512
#define HID     1024
#define OUT     512
#define G3      (3 * HID)          // 3072
#define SCAN_WG 96                 // 96 blocks * 8 waves; block b owns n in {2b,2b+1}
#define SCAN_THREADS 256
#define STEP_USHORTS (4 * 32 * 512)   // one step of packed h: 128 KB

typedef __attribute__((ext_vector_type(16))) __bf16 v16bf;
typedef __attribute__((ext_vector_type(8)))  float  v8f;

// exact parameter type expected by the async-to-LDS builtin (from hipcc diag)
typedef int vint4 __attribute__((vector_size(16)));
typedef __attribute__((address_space(1))) vint4* gvec_p;
typedef __attribute__((address_space(3))) vint4* lvec_p;

#define WMMA_BF16(a, b, c) \
    __builtin_amdgcn_wmma_f32_16x16x32_bf16(false, (a), false, (b), (short)0, (c), false, false)

// float -> bf16 (round-to-nearest-even)
__device__ __forceinline__ unsigned short f2bf(float f) {
    unsigned u = __float_as_uint(f);
    u += 0x7fffu + ((u >> 16) & 1u);
    return (unsigned short)(u >> 16);
}

// Load one 32B-per-lane fragment (global or LDS; addrspace inferred)
__device__ __forceinline__ v16bf load_frag(const unsigned short* tile, int lane) {
    union { uint4 q[2]; v16bf v; } u;
    const uint4* p = (const uint4*)tile + lane * 2;
    u.q[0] = p[0];
    u.q[1] = p[1];
    return u.v;
}

// 16B global -> LDS copy, async if the CDNA5 builtin exists
__device__ __forceinline__ void cp16_g2l(const unsigned short* __restrict__ g,
                                         unsigned short* l) {
#if __has_builtin(__builtin_amdgcn_global_load_async_to_lds_b128)
    __builtin_amdgcn_global_load_async_to_lds_b128((gvec_p)g, (lvec_p)l, 0, 0);
#else
    *(uint4*)l = *(const uint4*)g;
#endif
}

__device__ __forceinline__ void wait_async_copies() {
#if __has_builtin(__builtin_amdgcn_global_load_async_to_lds_b128)
#if __has_builtin(__builtin_amdgcn_s_wait_asynccnt)
    __builtin_amdgcn_s_wait_asynccnt(0);
#else
    asm volatile("s_wait_asynccnt 0" ::: "memory");
#endif
#endif
}

// --------------------------- packing kernels -------------------------------
// A-matrix 16x32 bf16 layout (ISA 7.12.2): lane = hi*16 + row,
//   j<4 : K = hi*8 + 2j + p ;  j>=4 : K = 16 + hi*8 + 2(j-4) + p
__device__ __forceinline__ void a_lane_slot(int q, int rr, int& lane, int& slot) {
    int hi, jj, p;
    if (q < 16) { hi = (q >= 8); int qq = q - hi * 8; jj = qq >> 1; p = qq & 1; }
    else        { int q2 = q - 16; hi = (q2 >= 8); int qq = q2 - hi * 8; jj = 4 + (qq >> 1); p = qq & 1; }
    lane = hi * 16 + rr;
    slot = 2 * jj + p;
}

__global__ void pack_A(const float* __restrict__ src, unsigned short* __restrict__ dst,
                       int R, int C) {
    long e = (long)blockIdx.x * blockDim.x + threadIdx.x;
    long total = (long)R * C;
    if (e >= total) return;
    int row = (int)(e / C), col = (int)(e % C);
    int mt = row >> 4, rr = row & 15, kt = col >> 5, q = col & 31;
    int lane, slot;
    a_lane_slot(q, rr, lane, slot);
    int Kt = C >> 5;
    dst[((size_t)(mt * Kt + kt)) * 512 + lane * 16 + slot] = f2bf(src[e]);
}

// B-matrix 32x16 bf16 layout: lane = hi*16 + n, slot<8: K = hi*8 + slot ; slot>=8: K = 16 + hi*8 + (slot-8)
__global__ void pack_B(const float* __restrict__ src, unsigned short* __restrict__ dst,
                       int K, int N) {
    long e = (long)blockIdx.x * blockDim.x + threadIdx.x;
    long total = (long)K * N;
    if (e >= total) return;
    int krow = (int)(e / N), ncol = (int)(e % N);
    int kt = krow >> 5, q = krow & 31, nt = ncol >> 4, n0 = ncol & 15;
    int hi = (q >> 3) & 1, rem = (q & 7) + 8 * (q >> 4), jj = rem >> 1, p = rem & 1;
    int lane = hi * 16 + n0, slot = 2 * jj + p;
    int Kt = K >> 5;
    dst[((size_t)(nt * Kt + kt)) * 512 + lane * 16 + slot] = f2bf(src[e]);
}

// ------------------------------ init ---------------------------------------
__global__ void init_ws(unsigned* __restrict__ bar, float* __restrict__ h,
                        unsigned short* __restrict__ zero_pack) {
    int i = blockIdx.x * blockDim.x + threadIdx.x;
    if (i < 64) bar[i] = 0u;
    if (i < BATCH * HID) h[i] = 0.f;
    if (i < STEP_USHORTS) zero_pack[i] = 0;
}

// --------------------------- bulk WMMA GEMM --------------------------------
// Each wave: 1 M-tile x 4 N-tiles (A fragment reused 4x).
__global__ void gemm_bias(const unsigned short* __restrict__ Apack,
                          const unsigned short* __restrict__ Bpack,
                          const float* __restrict__ bias,
                          float* __restrict__ out,
                          int Mtiles, int Ngroups, int Ktiles, int ldOut) {
    int gw   = (blockIdx.x * blockDim.x + threadIdx.x) >> 5;
    int lane = threadIdx.x & 31;
    if (gw >= Mtiles * Ngroups) return;        // wave-uniform exit
    int mt = gw / Ngroups, ng = gw % Ngroups;
    const unsigned short* a = Apack + (size_t)mt * Ktiles * 512;
    const unsigned short* b = Bpack + (size_t)(ng * 4) * Ktiles * 512;
    v8f c0 = {}, c1 = {}, c2 = {}, c3 = {};
#pragma unroll 2
    for (int k = 0; k < Ktiles; ++k) {
        v16bf av = load_frag(a + k * 512, lane);
        v16bf bv0 = load_frag(b + (size_t)(0 * Ktiles + k) * 512, lane);
        v16bf bv1 = load_frag(b + (size_t)(1 * Ktiles + k) * 512, lane);
        v16bf bv2 = load_frag(b + (size_t)(2 * Ktiles + k) * 512, lane);
        v16bf bv3 = load_frag(b + (size_t)(3 * Ktiles + k) * 512, lane);
        c0 = WMMA_BF16(av, bv0, c0);
        c1 = WMMA_BF16(av, bv1, c1);
        c2 = WMMA_BF16(av, bv2, c2);
        c3 = WMMA_BF16(av, bv3, c3);
    }
    int hi = lane >> 4, n0 = lane & 15;
    v8f cc[4] = {c0, c1, c2, c3};
#pragma unroll
    for (int i = 0; i < 4; ++i) {
        int col = (ng * 4 + i) * 16 + n0;
        float bb = bias[col];
#pragma unroll
        for (int j = 0; j < 8; ++j) {
            int row = mt * 16 + hi * 8 + j;
            out[(size_t)row * ldOut + col] = cc[i][j] + bb;
        }
    }
}

// --------------------------- grid barrier ----------------------------------
__device__ __forceinline__ void grid_barrier(unsigned* cnt, unsigned target) {
    __syncthreads();
    if (threadIdx.x == 0) {
        __threadfence();
        __hip_atomic_fetch_add(cnt, 1u, __ATOMIC_RELEASE, __HIP_MEMORY_SCOPE_AGENT);
        while (__hip_atomic_load(cnt, __ATOMIC_ACQUIRE, __HIP_MEMORY_SCOPE_AGENT) < target) {
            __builtin_amdgcn_s_sleep(1);
        }
    }
    __syncthreads();
}

// --------------------------- persistent GRU scan ---------------------------
// Block b: waves w=0..7 -> (m = w&3, n = 2b + (w>>2)).
// LDS: Wh slices for the block's two n-tiles (64 KB, staged once) +
//      h_{t-1} packed tiles (128 KB, async-staged per step).
__global__ void __launch_bounds__(SCAN_THREADS, 1)
gru_scan(const float* __restrict__ xg,            // [T,B,3H] fp32
         const unsigned short* __restrict__ Whp,  // swizzled B tiles [192][32]
         const unsigned short* __restrict__ zpak, // zero A tiles (one step)
         const float* __restrict__ bh,            // [3H]
         unsigned short* __restrict__ hs_pack,    // [T][4*32] swizzled A tiles
         float* __restrict__ hg,                  // [B,3H] scratch
         float* __restrict__ h,                   // [B,H] running state fp32
         unsigned* __restrict__ bar) {
    __shared__ unsigned short Blds[2 * 32 * 512];      //  64 KB
    __shared__ unsigned short Alds[STEP_USHORTS];      // 128 KB

    const int lane = threadIdx.x & 31;
    const int wave = threadIdx.x >> 5;
    const int m  = wave & 3;
    const int nl = wave >> 2;                          // 0 or 1
    const int n  = blockIdx.x * 2 + nl;                // 0..191
    const int hi = lane >> 4;
    const int col = n * 16 + (lane & 15);
    const float bcol = bh[col];

    // ---- stage this block's Wh slices into LDS (once) ----
    {
        const unsigned short* src = Whp + (size_t)(blockIdx.x * 2) * 32 * 512;
        for (int i = threadIdx.x; i < (2 * 32 * 512) / 8; i += SCAN_THREADS)
            cp16_g2l(src + i * 8, &Blds[i * 8]);
        wait_async_copies();
        __syncthreads();
    }

    unsigned gen = 0;
    for (int t = 0; t < T_STEPS; ++t) {
        // ---- async-stage h_{t-1} packed tiles into LDS ----
        const unsigned short* Ap =
            (t == 0) ? zpak : (hs_pack + (size_t)(t - 1) * STEP_USHORTS);
        for (int i = threadIdx.x; i < STEP_USHORTS / 8; i += SCAN_THREADS)
            cp16_g2l(Ap + i * 8, &Alds[i * 8]);
        wait_async_copies();
        __syncthreads();

        // ---- hg tile = h_{t-1} @ Wh (bf16 WMMA from LDS, fp32 acc) ----
        const unsigned short* a = &Alds[(m * 32) * 512];
        const unsigned short* b = &Blds[(nl * 32) * 512];
        v8f c = {};
#pragma unroll 4
        for (int k = 0; k < 32; ++k) {
            v16bf av = load_frag(a + k * 512, lane);
            v16bf bv = load_frag(b + k * 512, lane);
            c = WMMA_BF16(av, bv, c);
        }
#pragma unroll
        for (int j = 0; j < 8; ++j)
            hg[(size_t)(m * 16 + hi * 8 + j) * G3 + col] = c[j] + bcol;

        ++gen;
        grid_barrier(bar, (unsigned)SCAN_WG * gen);

        // ---- gates + state update; write h_t as swizzled bf16 A-tiles ----
        const float* xgt = xg + (size_t)t * BATCH * G3;
        for (int e = blockIdx.x * SCAN_THREADS + threadIdx.x;
             e < BATCH * HID; e += SCAN_WG * SCAN_THREADS) {
            int bb = e >> 10, j = e & (HID - 1);
            size_t base = (size_t)bb * G3 + j;
            float r  = 1.f / (1.f + expf(-(xgt[base]              + hg[base])));
            float z  = 1.f / (1.f + expf(-(xgt[base + HID]        + hg[base + HID])));
            float nn = tanhf(xgt[base + 2 * HID] + r * hg[base + 2 * HID]);
            float hp = h[e];
            float hnew = (1.f - z) * nn + z * hp;
            h[e] = hnew;
            int mt = bb >> 4, rr = bb & 15, kt = j >> 5, q = j & 31;
            int lane2, slot;
            a_lane_slot(q, rr, lane2, slot);
            hs_pack[(size_t)t * STEP_USHORTS +
                    (size_t)(mt * 32 + kt) * 512 + lane2 * 16 + slot] = f2bf(hnew);
        }

        ++gen;
        grid_barrier(bar, (unsigned)SCAN_WG * gen);
    }
}

// ----------------------------- host launcher -------------------------------
extern "C" void kernel_launch(void* const* d_in, const int* in_sizes, int n_in,
                              void* d_out, int out_size, void* d_ws, size_t ws_size,
                              hipStream_t stream) {
    const float* x  = (const float*)d_in[0];  // [T,B,D_IN]
    const float* Wi = (const float*)d_in[1];  // [D_IN,3H]
    const float* Wh = (const float*)d_in[2];  // [H,3H]
    const float* bi = (const float*)d_in[3];  // [3H]
    const float* bh = (const float*)d_in[4];  // [3H]
    const float* Wo = (const float*)d_in[5];  // [H,O]
    const float* bo = (const float*)d_in[6];  // [O]
    float* y = (float*)d_out;                 // [T,B,O]

    char* w = (char*)d_ws;
    size_t off = 0;
    auto alloc = [&](size_t bytes) { void* p = w + off; off = (off + bytes + 255) & ~(size_t)255; return p; };

    unsigned*       bar       = (unsigned*)alloc(256);
    float*          h_state   = (float*)alloc((size_t)BATCH * HID * 4);
    unsigned short* zero_pack = (unsigned short*)alloc((size_t)STEP_USHORTS * 2);
    float*          hg        = (float*)alloc((size_t)BATCH * G3 * 4);
    unsigned short* x_pack    = (unsigned short*)alloc((size_t)T_STEPS * BATCH * D_IN * 2);
    unsigned short* Wi_pack   = (unsigned short*)alloc((size_t)D_IN * G3 * 2);
    unsigned short* Wh_pack   = (unsigned short*)alloc((size_t)HID * G3 * 2);
    unsigned short* Wo_pack   = (unsigned short*)alloc((size_t)HID * OUT * 2);
    float*          xg        = (float*)alloc((size_t)T_STEPS * BATCH * G3 * 4);
    unsigned short* hs_pack   = (unsigned short*)alloc((size_t)T_STEPS * STEP_USHORTS * 2);
    (void)ws_size; (void)in_sizes; (void)n_in; (void)out_size;

    // 0) init barrier counters, h0 = 0, zero A-tiles
    init_ws<<<(BATCH * HID + 255) / 256, 256, 0, stream>>>(bar, h_state, zero_pack);

    // 1) pack operands into WMMA-fragment-swizzled bf16
    {
        long n = (long)T_STEPS * BATCH * D_IN;
        pack_A<<<(unsigned)((n + 255) / 256), 256, 0, stream>>>(x, x_pack, T_STEPS * BATCH, D_IN);
    }
    {
        long n = (long)D_IN * G3;
        pack_B<<<(unsigned)((n + 255) / 256), 256, 0, stream>>>(Wi, Wi_pack, D_IN, G3);
    }
    {
        long n = (long)HID * G3;
        pack_B<<<(unsigned)((n + 255) / 256), 256, 0, stream>>>(Wh, Wh_pack, HID, G3);
    }
    {
        long n = (long)HID * OUT;
        pack_B<<<(unsigned)((n + 255) / 256), 256, 0, stream>>>(Wo, Wo_pack, HID, OUT);
    }

    // 2) xg = x @ Wi + bi   (Mtiles=2048, Ngroups=48, Ktiles=16)
    {
        int Mt = (T_STEPS * BATCH) / 16, Ng = (G3 / 16) / 4, Kt = D_IN / 32;
        int waves = Mt * Ng;
        gemm_bias<<<(waves + 7) / 8, 256, 0, stream>>>(x_pack, Wi_pack, bi, xg, Mt, Ng, Kt, G3);
    }

    // 3) persistent recurrent scan
    gru_scan<<<SCAN_WG, SCAN_THREADS, 0, stream>>>(xg, Wh_pack, zero_pack, bh,
                                                   hs_pack, hg, h_state, bar);

    // 4) y = hs @ Wo + bo   (Mtiles=2048, Ngroups=8, Ktiles=32)
    {
        int Mt = (T_STEPS * BATCH) / 16, Ng = (OUT / 16) / 4, Kt = HID / 32;
        int waves = Mt * Ng;
        gemm_bias<<<(waves + 7) / 8, 256, 0, stream>>>(hs_pack, Wo_pack, bo, y, Mt, Ng, Kt, OUT);
    }
}
